// t2_quantizedCNN_5291399709313
// MI455X (gfx1250) — 8-bit matrix operand order repaired and checked by reading, compiled, not run on hardware
//
#include <hip/hip_runtime.h>


typedef __attribute__((ext_vector_type(8))) int v8i;

#define TILE 16
#define NTHR 256

__device__ __forceinline__ int dot4_acc(unsigned int a, unsigned int b, int acc) {
#if __has_builtin(__builtin_amdgcn_sdot4)
  return __builtin_amdgcn_sdot4((int)a, (int)b, acc, false);
#elif __has_builtin(__builtin_amdgcn_sudot4)
  return __builtin_amdgcn_sudot4(true, (int)b, false, (int)a, acc, false);
#else
  acc += (int)(signed char)(a      ) * (int)(signed char)(b      );
  acc += (int)(signed char)(a >>  8) * (int)(signed char)(b >>  8);
  acc += (int)(signed char)(a >> 16) * (int)(signed char)(b >> 16);
  acc += (int)(signed char)(a >> 24) * (int)(signed char)(b >> 24);
  return acc;
#endif
}

__device__ __forceinline__ float wq(float w) {
  return fminf(fmaxf(rintf(w), -1.0f), 1.0f);
}

__device__ __forceinline__ unsigned char aq(float v, float s) {
  float r = rintf(fmaxf(v, 0.0f) / s);
  return (unsigned char)fminf(fmaxf(r, 0.0f), 3.0f);
}

__global__ __launch_bounds__(NTHR)
void qcnn_fused(const float* __restrict__ x,
                const float* __restrict__ w1, const float* __restrict__ w2,
                const float* __restrict__ w3, const float* __restrict__ w4,
                const float* __restrict__ wfc,
                const float* __restrict__ s1p, const float* __restrict__ s2p,
                const float* __restrict__ s3p, const float* __restrict__ s4p,
                float* __restrict__ out, int nimg)
{
  __shared__ float        xin[4 * 32 * 32];
  __shared__ unsigned int h1w[16 * 16 * 4];
  __shared__ unsigned int h2w[8 * 8 * 2];
  __shared__ unsigned int h3w[4 * 4 * 4];
  __shared__ float        w1qf[16 * 4 * 9];
  __shared__ unsigned int w2qw[8 * 9 * 4];
  __shared__ unsigned int w3qw[16 * 9 * 2];
  __shared__ unsigned int w4qw[8 * 9 * 4];
  __shared__ unsigned int featw[TILE * 8];
  __shared__ float        sv[4];

  const int tid = threadIdx.x;
  const int b0  = blockIdx.x * TILE;

  for (int i = tid; i < 576; i += NTHR) w1qf[i] = wq(w1[i]);
  {
    signed char* w2b = (signed char*)w2qw;
    for (int i = tid; i < 1152; i += NTHR) {
      int oc = i / 144, ic = (i / 9) % 16, t = i % 9;
      w2b[(oc * 9 + t) * 16 + ic] = (signed char)(int)wq(w2[i]);
    }
    signed char* w3b = (signed char*)w3qw;
    for (int i = tid; i < 1152; i += NTHR) {
      int oc = i / 72, ic = (i / 9) % 8, t = i % 9;
      w3b[(oc * 9 + t) * 8 + ic] = (signed char)(int)wq(w3[i]);
    }
    signed char* w4b = (signed char*)w4qw;
    for (int i = tid; i < 1152; i += NTHR) {
      int oc = i / 144, ic = (i / 9) % 16, t = i % 9;
      w4b[(oc * 9 + t) * 16 + ic] = (signed char)(int)wq(w4[i]);
    }
  }
  if (tid == 0) { sv[0] = s1p[0]; sv[1] = s2p[0]; sv[2] = s3p[0]; sv[3] = s4p[0]; }
  if (tid < TILE * 8) featw[tid] = 0u;
  __syncthreads();

  for (int img = 0; img < TILE; ++img) {
    const int b = b0 + img;
    if (b >= nimg) break;

    const float* xp = x + (size_t)b * 4096;
    {
      const float4* xs = (const float4*)xp;
      float4*       xd = (float4*)xin;
      for (int i = tid; i < 1024; i += NTHR) xd[i] = xs[i];
      if (b + 1 < nimg)
        __builtin_prefetch((const char*)(xp + 4096) + tid * 64, 0, 1);
    }
    __syncthreads();

    for (int i = tid; i < 4096; i += NTHR) {
      const int oc = i & 15, ox = (i >> 4) & 15, oy = i >> 8;
      const int iy0 = 2 * oy - 1, ix0 = 2 * ox - 1;
      const float* wp = &w1qf[oc * 36];
      float acc = 0.0f;
#pragma unroll
      for (int ky = 0; ky < 3; ++ky) {
        const int iy = iy0 + ky;
        if ((unsigned)iy < 32u) {
#pragma unroll
          for (int kx = 0; kx < 3; ++kx) {
            const int ix = ix0 + kx;
            if ((unsigned)ix < 32u) {
              const int p = iy * 32 + ix, t = ky * 3 + kx;
              acc += xin[p]        * wp[t];
              acc += xin[1024 + p] * wp[9 + t];
              acc += xin[2048 + p] * wp[18 + t];
              acc += xin[3072 + p] * wp[27 + t];
            }
          }
        }
      }
      ((unsigned char*)h1w)[(oy * 16 + ox) * 16 + oc] = aq(acc, sv[0]);
    }
    __syncthreads();

    for (int i = tid; i < 512; i += NTHR) {
      const int oc = i & 7, ox = (i >> 3) & 7, oy = i >> 6;
      int acc = 0;
#pragma unroll
      for (int ky = 0; ky < 3; ++ky) {
        const int iy = 2 * oy - 1 + ky;
        if ((unsigned)iy < 16u) {
#pragma unroll
          for (int kx = 0; kx < 3; ++kx) {
            const int ix = 2 * ox - 1 + kx;
            if ((unsigned)ix < 16u) {
              const unsigned int* hp = &h1w[(iy * 16 + ix) * 4];
              const unsigned int* wv = &w2qw[(oc * 9 + ky * 3 + kx) * 4];
              acc = dot4_acc(hp[0], wv[0], acc);
              acc = dot4_acc(hp[1], wv[1], acc);
              acc = dot4_acc(hp[2], wv[2], acc);
              acc = dot4_acc(hp[3], wv[3], acc);
            }
          }
        }
      }
      ((unsigned char*)h2w)[(oy * 8 + ox) * 8 + oc] = aq((float)acc * sv[0], sv[1]);
    }
    __syncthreads();

    {
      const int oc = tid & 15, ox = (tid >> 4) & 3, oy = tid >> 6;
      int acc = 0;
#pragma unroll
      for (int ky = 0; ky < 3; ++ky) {
        const int iy = 2 * oy - 1 + ky;
        if ((unsigned)iy < 8u) {
#pragma unroll
          for (int kx = 0; kx < 3; ++kx) {
            const int ix = 2 * ox - 1 + kx;
            if ((unsigned)ix < 8u) {
              const unsigned int* hp = &h2w[(iy * 8 + ix) * 2];
              const unsigned int* wv = &w3qw[(oc * 9 + ky * 3 + kx) * 2];
              acc = dot4_acc(hp[0], wv[0], acc);
              acc = dot4_acc(hp[1], wv[1], acc);
            }
          }
        }
      }
      ((unsigned char*)h3w)[(oy * 4 + ox) * 16 + oc] = aq((float)acc * sv[1], sv[2]);
    }
    __syncthreads();

    if (tid < 32) {
      const int oc = tid & 7, ox = (tid >> 3) & 1, oy = tid >> 4;
      int acc = 0;
#pragma unroll
      for (int ky = 0; ky < 3; ++ky) {
        const int iy = 2 * oy - 1 + ky;
        if ((unsigned)iy < 4u) {
#pragma unroll
          for (int kx = 0; kx < 3; ++kx) {
            const int ix = 2 * ox - 1 + kx;
            if ((unsigned)ix < 4u) {
              const unsigned int* hp = &h3w[(iy * 4 + ix) * 4];
              const unsigned int* wv = &w4qw[(oc * 9 + ky * 3 + kx) * 4];
              acc = dot4_acc(hp[0], wv[0], acc);
              acc = dot4_acc(hp[1], wv[1], acc);
              acc = dot4_acc(hp[2], wv[2], acc);
              acc = dot4_acc(hp[3], wv[3], acc);
            }
          }
        }
      }
      ((unsigned char*)featw)[img * 32 + oc * 4 + oy * 2 + ox] =
          aq((float)acc * sv[2], sv[3]);
    }
    __syncthreads();
  }

  {
    const int lane = tid & 31;
    const int m    = lane & 15;
    const int hi   = lane >> 4;
    v8i a, bm;
    v8i c = {};

#pragma unroll
    for (int v = 0; v < 8; ++v) {
      const int k0 = (v >> 2) * 32 + hi * 16 + (v & 3) * 4;
      a[v] = (k0 < 32) ? (int)featw[m * 8 + (k0 >> 2)] : 0;
    }

#pragma unroll
    for (int v = 0; v < 8; ++v) {
      const int k0 = (v >> 2) * 32 + hi * 16 + (v & 3) * 4;
      int word = 0;
      if (m < 6 && k0 < 32) {
#pragma unroll
        for (int j = 0; j < 4; ++j) {
          const int q = (int)wq(wfc[m * 32 + k0 + j]);
          word |= (q & 0xff) << (8 * j);
        }
      }
      bm[v] = word;
    }

    v8i d = __builtin_amdgcn_wmma_i32_16x16x64_iu8(false, a, true, bm, c,
                                                   false, false);

    const float s4 = sv[3];
    if (tid < 32 && m < 6) {
#pragma unroll
      for (int g = 0; g < 8; ++g) {
        const int row = b0 + hi * 8 + g;
        if (row < nimg) out[row * 6 + m] = s4 * (float)d[g];
      }
    }
  }
}

extern "C" void kernel_launch(void* const* d_in, const int* in_sizes, int n_in,
                              void* d_out, int out_size, void* d_ws, size_t ws_size,
                              hipStream_t stream) {
  (void)n_in; (void)d_ws; (void)ws_size; (void)out_size;
  const float* x   = (const float*)d_in[0];
  const float* w1  = (const float*)d_in[1];
  const float* w2  = (const float*)d_in[2];
  const float* w3  = (const float*)d_in[3];
  const float* w4  = (const float*)d_in[4];
  const float* wfc = (const float*)d_in[5];
  const float* s1  = (const float*)d_in[6];
  const float* s2  = (const float*)d_in[7];
  const float* s3  = (const float*)d_in[8];
  const float* s4  = (const float*)d_in[9];

  const int nimg   = in_sizes[0] / (4 * 32 * 32);
  const int blocks = (nimg + TILE - 1) / TILE;

  qcnn_fused<<<blocks, NTHR, 0, stream>>>(x, w1, w2, w3, w4, wfc,
                                          s1, s2, s3, s4,
                                          (float*)d_out, nimg);
}
